// MultiHeadSelfAttention_Pytorch_50483045597400
// MI455X (gfx1250) — matrix operand order repaired and checked by reading, not compiled, not run on hardware
//
#include <hip/hip_runtime.h>

// ---------------------------------------------------------------------------
// MHA forward for MI455X (gfx1250, wave32, WMMA + Tensor Data Mover).
// fp32 inputs -> bf16 hi/lo split, 3-term WMMA emulation (~fp32 accuracy at
// ~2.7x the native f32-WMMA rate). GEMMs on v_wmma_f32_16x16x32_bf16.
// Weight tiles streamed into LDS by per-wave TDM DMAs (tensor_load_to_lds,
// TENSORcnt, double-buffered) -- fully uniform control flow, vector-memory
// pipe stays free for A-operands.
// ---------------------------------------------------------------------------

typedef __attribute__((ext_vector_type(16))) __bf16 v16bf;
typedef __attribute__((ext_vector_type(8)))  __bf16 v8bf;
typedef __attribute__((ext_vector_type(8)))  float  v8f;
typedef __attribute__((ext_vector_type(4)))  unsigned int v4u;
typedef __attribute__((ext_vector_type(8)))  int    v8i;
typedef __attribute__((ext_vector_type(4)))  int    v4i;

#define DEV static __device__ __forceinline__

constexpr int Bsz = 2;
constexpr int S   = 2048;
constexpr int D   = 1024;
constexpr int H   = 16;
constexpr int DK  = 64;
constexpr int M   = Bsz * S;      // 4096 flattened rows
constexpr int WPITCH = 48;        // LDS row pitch (elems): 64B row + 32B TDM pad

DEV v8f wmma_bf16(v16bf a, v16bf b, v8f c) {
  // D = A(16x32) * B(32x16) + C, f32 accumulate
  return __builtin_amdgcn_wmma_f32_16x16x32_bf16(
      /*neg_a=*/false, a, /*neg_b=*/false, b,
      /*c_mod=*/(short)0, c, /*reuse_a=*/false, /*reuse_b=*/false);
}

DEV v16bf load_pair16(const __bf16* p0, const __bf16* p1) {
  // A-operand gather: two contiguous 16B chunks -> v16bf
  v8bf a = *(const v8bf*)p0;
  v8bf b = *(const v8bf*)p1;
  v16bf r;
#pragma unroll
  for (int i = 0; i < 8; ++i) { r[i] = a[i]; r[i + 8] = b[i]; }
  return r;
}

DEV v8f zero8() { v8f z = {0.f,0.f,0.f,0.f,0.f,0.f,0.f,0.f}; return z; }

DEV void split1(float x, __bf16& h, __bf16& l) {
  h = (__bf16)x;
  l = (__bf16)(x - (float)h);
}

// ---- Tensor Data Mover: DMA one 16x32 bf16 tile (row stride D) into LDS ----
// Pad mode: +32B after every 64B row -> LDS pitch 96B (conflict-free, 32B
// aligned v16bf reads).  D# built per CDNA5 ISA.
// Each wave DMAs its own quarter of the block tile (TDM ignores EXEC,
// TENSORcnt is per-wave -> no divergent control flow needed).
DEV void tdm_load_tile16(void* lds_dst, const void* gsrc) {
  unsigned lds = (unsigned)(unsigned long long)lds_dst;   // aperture: low 32b = LDS offset
  unsigned long long ga = (unsigned long long)gsrc;
  v4u g0;
  g0[0] = 1u;                                             // count=1, user D#
  g0[1] = lds;                                            // lds_addr
  g0[2] = (unsigned)(ga & 0xFFFFFFFFu);                   // global_addr[31:0]
  g0[3] = (unsigned)((ga >> 32) & 0x01FFFFFFu) | 0x80000000u; // addr[56:32] | type=2
  v8i g1;
  // [17:16] data_size=1(2B) | [20] pad_en | [24:22] pad_interval=3 (16 DW)
  // | [31:25] pad_amount=7 (8 DW)
  g1[0] = (int)((1u << 16) | (1u << 20) | (3u << 22) | (7u << 25));
  g1[1] = (int)((unsigned)(D & 0xFFFF) << 16);            // tensor_dim0 lo16 (1024)
  g1[2] = (int)((unsigned)(D & 0xFFFF) << 16);            // dim0 hi16=0 | tensor_dim1 lo16
  g1[3] = (int)(32u << 16);                               // dim1 hi16=0 | tile_dim0=32
  g1[4] = 16;                                             // tile_dim1=16 | tile_dim2=0
  g1[5] = D;                                              // tensor_dim0_stride lo32
  g1[6] = 0;                                              // stride hi | dim1_stride lo
  g1[7] = 0;
  v4i z4 = {0, 0, 0, 0};                                  // 2D tensor: groups 2/3 unused
#if defined(__clang_major__) && (__clang_major__ >= 23)
  v8i z8 = {0, 0, 0, 0, 0, 0, 0, 0};
  __builtin_amdgcn_tensor_load_to_lds(g0, g1, z4, z4, z8, 0);
#else
  __builtin_amdgcn_tensor_load_to_lds(g0, g1, z4, z4, 0);
#endif
}

// --------------------------- split fp32 -> bf16 hi/lo ----------------------
__global__ void split_f32_kernel(const float* __restrict__ src,
                                 __bf16* __restrict__ hi,
                                 __bf16* __restrict__ lo, int n) {
  int i = blockIdx.x * blockDim.x + threadIdx.x;
  if (i < n) {
    __bf16 h, l;
    split1(src[i], h, l);
    hi[i] = h; lo[i] = l;
  }
}

// --------------------------- QKV projection GEMM ---------------------------
// out[m,n] = sum_k X[m,k] * W[n,k] + bias[n]   (torch Linear x @ W^T + b)
// Block = 4 waves sharing one 64-wide N strip (= one head); each wave TDM-DMAs
// a 16-row quarter of the W tile into LDS (double-buffered); each wave owns a
// 16-row M tile.  Output stored as bf16 hi/lo:
//   transposed==0 (q,k): [bh][s][dk]   transposed==1 (v): [bh][dk][s]
__global__ __launch_bounds__(128) void qkv_proj_kernel(
    const __bf16* __restrict__ Xhi, const __bf16* __restrict__ Xlo,
    const __bf16* __restrict__ Whi, const __bf16* __restrict__ Wlo,
    const float*  __restrict__ bias,
    __bf16* __restrict__ outHi, __bf16* __restrict__ outLo,
    int transposed) {
  __shared__ alignas(32) __bf16 Wt[2][2][64 * WPITCH];  // [buf][hi/lo][row*pitch]

  const int lane = threadIdx.x & 31;
  const int wid  = threadIdx.x >> 5;
  const int half = lane >> 4;
  const int lm   = lane & 15;
  const int n0   = blockIdx.x * 64;             // 64-wide N strip == one head
  const int m0   = blockIdx.y * 64 + wid * 16;  // 16-row M tile per wave

  v8f acc[4];
#pragma unroll
  for (int t = 0; t < 4; ++t) acc[t] = zero8();

  const __bf16* pa  = Xhi + (m0 + lm) * D;
  const __bf16* pal = Xlo + (m0 + lm) * D;
  // this wave's quarter of the weight strip (16 rows)
  const __bf16* wh0 = Whi + (size_t)(n0 + wid * 16) * D;
  const __bf16* wl0 = Wlo + (size_t)(n0 + wid * 16) * D;
  const int ldsrow  = wid * 16 * WPITCH;

  // prologue: every wave DMAs its quarter of buffer 0 (uniform, no branch)
  tdm_load_tile16(&Wt[0][0][ldsrow], wh0);
  tdm_load_tile16(&Wt[0][1][ldsrow], wl0);

  for (int k0 = 0; k0 < D; k0 += 32) {
    const int cur = (k0 >> 5) & 1;
    if (k0 + 32 < D) {                           // uniform scalar condition
      tdm_load_tile16(&Wt[cur ^ 1][0][ldsrow], wh0 + k0 + 32);
      tdm_load_tile16(&Wt[cur ^ 1][1][ldsrow], wl0 + k0 + 32);
      __builtin_amdgcn_s_wait_tensorcnt(2);      // this wave's current pair done
    } else {
      __builtin_amdgcn_s_wait_tensorcnt(0);
    }
    __syncthreads();                             // publish LDS tile to all waves

    v16bf ahi = load_pair16(pa  + k0 + half * 16, pa  + k0 + half * 16 + 8);
    v16bf alo = load_pair16(pal + k0 + half * 16, pal + k0 + half * 16 + 8);
#pragma unroll
    for (int t = 0; t < 4; ++t) {
      const int row = t * 16 + lm;               // B-lane = output feature row
      v16bf bhi = *(const v16bf*)&Wt[cur][0][row * WPITCH + half * 16];
      v16bf blo = *(const v16bf*)&Wt[cur][1][row * WPITCH + half * 16];
      acc[t] = wmma_bf16(alo, bhi, acc[t]);      // low-order terms first
      acc[t] = wmma_bf16(ahi, blo, acc[t]);
      acc[t] = wmma_bf16(ahi, bhi, acc[t]);
    }
    __syncthreads();                             // tile consumed; buffer reusable
  }

  const int h = n0 >> 6;                         // head id (strip aligned)
#pragma unroll
  for (int t = 0; t < 4; ++t) {
    const int ng = n0 + t * 16 + lm;
    const int dk = t * 16 + lm;                  // ng & 63
    const float bv = bias[ng];
#pragma unroll
    for (int r = 0; r < 8; ++r) {
      const int mg = m0 + r + 8 * half;          // row in [0, 4096)
      const int b  = mg >> 11;
      const int s  = mg & (S - 1);
      float v = acc[t][r] + bv;
      __bf16 vh, vl;
      split1(v, vh, vl);
      int idx;
      if (!transposed) idx = ((b * H + h) * S + s) * DK + dk;
      else             idx = ((b * H + h) * DK + dk) * S + s;
      outHi[idx] = vh;
      outLo[idx] = vl;
    }
  }
}

// ------------------------------- attention ---------------------------------
// Per wave: one 16-row query tile over all 2048 keys for one (b,h).
// scores = exp(q.k^T / 8) (raw exp, per reference); rowsum via WMMA-with-ones
// so normalization is lane-local; context accumulated over transposed V.
__global__ __launch_bounds__(128) void attn_kernel(
    const __bf16* __restrict__ qhi, const __bf16* __restrict__ qlo,
    const __bf16* __restrict__ khi, const __bf16* __restrict__ klo,
    const __bf16* __restrict__ vthi, const __bf16* __restrict__ vtlo,
    float* __restrict__ out) {
  __shared__ float Pbuf[4][16][36];              // per-wave P32 staging tile

  const int lane = threadIdx.x & 31;
  const int wid  = threadIdx.x >> 5;
  const int half = lane >> 4;
  const int lm   = lane & 15;
  const int bh   = blockIdx.y;
  const int m0   = (blockIdx.x * 4 + wid) * 16;  // query rows within this bh

  // ones matrix for rowsum-by-WMMA
  v16bf ones;
#pragma unroll
  for (int i = 0; i < 16; ++i) ones[i] = (__bf16)1.0f;

  // Preload q tile A-operands for both K-steps (d0 = 0, 32)
  const __bf16* qrh = qhi + (bh * S + m0 + lm) * DK;
  const __bf16* qrl = qlo + (bh * S + m0 + lm) * DK;
  v16bf Ahi[2], Alo[2];
#pragma unroll
  for (int ks = 0; ks < 2; ++ks) {
    const int d0 = ks * 32;
    Ahi[ks] = load_pair16(qrh + d0 + half * 16, qrh + d0 + half * 16 + 8);
    Alo[ks] = load_pair16(qrl + d0 + half * 16, qrl + d0 + half * 16 + 8);
  }

  v8f rs = zero8();
  v8f ctx[4];
#pragma unroll
  for (int t = 0; t < 4; ++t) ctx[t] = zero8();

  float (*P)[36] = Pbuf[wid];

  for (int t0 = 0; t0 < S; t0 += 32) {
    // ---- scores + exp for two 16-key sub-tiles -> P32 in LDS ----
#pragma unroll
    for (int sub = 0; sub < 2; ++sub) {
      const int trow = t0 + sub * 16 + lm;       // B-lane = key index
      const __bf16* kb  = khi + (bh * S + trow) * DK;
      const __bf16* kbl = klo + (bh * S + trow) * DK;
      v8f sc = zero8();
#pragma unroll
      for (int ks = 0; ks < 2; ++ks) {
        v16bf bhiV = *(const v16bf*)(kb  + ks * 32 + half * 16);
        v16bf bloV = *(const v16bf*)(kbl + ks * 32 + half * 16);
        sc = wmma_bf16(Alo[ks], bhiV, sc);
        sc = wmma_bf16(Ahi[ks], bloV, sc);
        sc = wmma_bf16(Ahi[ks], bhiV, sc);
      }
#pragma unroll
      for (int r = 0; r < 8; ++r)
        P[r + 8 * half][sub * 16 + lm] = __expf(sc[r] * 0.125f);
    }

    // ---- re-read P32 in A-layout (DS is in-order within a wave) ----
    v16bf phi, plo;
#pragma unroll
    for (int j = 0; j < 8; ++j) {
      float f0 = P[lm][half * 16 + j];
      float f1 = P[lm][half * 16 + 8 + j];
      __bf16 h0, l0, h1, l1;
      split1(f0, h0, l0);
      split1(f1, h1, l1);
      phi[j] = h0; plo[j] = l0;
      phi[j + 8] = h1; plo[j + 8] = l1;
    }

    // rowsum: B = ones -> every D column holds the row sum (lane-local later)
    rs = wmma_bf16(plo, ones, rs);
    rs = wmma_bf16(phi, ones, rs);

    // context: B = V^T rows are contiguous in key index
#pragma unroll
    for (int dt = 0; dt < 4; ++dt) {
      const int dkl = dt * 16 + lm;              // B-lane = head dim
      const __bf16* vb  = vthi + (bh * DK + dkl) * S + t0 + half * 16;
      const __bf16* vbl = vtlo + (bh * DK + dkl) * S + t0 + half * 16;
      v16bf bh16 = *(const v16bf*)vb;
      v16bf bl16 = *(const v16bf*)vbl;
      ctx[dt] = wmma_bf16(plo, bh16, ctx[dt]);
      ctx[dt] = wmma_bf16(phi, bl16, ctx[dt]);
      ctx[dt] = wmma_bf16(phi, bh16, ctx[dt]);
    }
  }

  // ---- normalize (rowsum sits in the same lane/VGPR slots) + store ----
  const int b = bh >> 4, h = bh & (H - 1);
#pragma unroll
  for (int r = 0; r < 8; ++r) {
    const float inv = 1.0f / (rs[r] + 1e-8f);
    const int s = m0 + r + 8 * half;
    float* orow = out + (b * S + s) * D + h * DK;
#pragma unroll
    for (int dt = 0; dt < 4; ++dt)
      orow[dt * 16 + lm] = ctx[dt][r] * inv;
  }
}

// ------------------------------- launcher ----------------------------------
extern "C" void kernel_launch(void* const* d_in, const int* in_sizes, int n_in,
                              void* d_out, int out_size, void* d_ws, size_t ws_size,
                              hipStream_t stream) {
  (void)in_sizes; (void)n_in; (void)out_size; (void)ws_size;
  const float* Qin = (const float*)d_in[0];
  const float* Wq  = (const float*)d_in[1];
  const float* bq  = (const float*)d_in[2];
  const float* Wk  = (const float*)d_in[3];
  const float* bk  = (const float*)d_in[4];
  const float* Wv  = (const float*)d_in[5];
  const float* bv  = (const float*)d_in[6];
  float* out = (float*)d_out;

  char* p = (char*)d_ws;
  auto take = [&](size_t elems) { __bf16* r = (__bf16*)p; p += elems * sizeof(__bf16); return r; };

  const size_t NX = (size_t)M * D;            // 4M elems
  const size_t NW = (size_t)D * D;            // 1M elems
  const size_t NQ = (size_t)Bsz * H * S * DK; // 4M elems

  __bf16* Xhi = take(NX);  __bf16* Xlo = take(NX);
  __bf16* Wqh = take(NW);  __bf16* Wql = take(NW);
  __bf16* Wkh = take(NW);  __bf16* Wkl = take(NW);
  __bf16* Wvh = take(NW);  __bf16* Wvl = take(NW);
  __bf16* qh  = take(NQ);  __bf16* ql  = take(NQ);
  __bf16* kh  = take(NQ);  __bf16* kl  = take(NQ);
  __bf16* vth = take(NQ);  __bf16* vtl = take(NQ);

  // 1) split inputs to bf16 hi/lo
  split_f32_kernel<<<(int)((NX + 255) / 256), 256, 0, stream>>>(Qin, Xhi, Xlo, (int)NX);
  split_f32_kernel<<<(int)((NW + 255) / 256), 256, 0, stream>>>(Wq,  Wqh, Wql, (int)NW);
  split_f32_kernel<<<(int)((NW + 255) / 256), 256, 0, stream>>>(Wk,  Wkh, Wkl, (int)NW);
  split_f32_kernel<<<(int)((NW + 255) / 256), 256, 0, stream>>>(Wv,  Wvh, Wvl, (int)NW);

  // 2) QKV projections (v stored transposed per head for the P.V GEMM)
  dim3 pgrid(D / 64, M / 64);
  qkv_proj_kernel<<<pgrid, 128, 0, stream>>>(Xhi, Xlo, Wqh, Wql, bq, qh,  ql,  0);
  qkv_proj_kernel<<<pgrid, 128, 0, stream>>>(Xhi, Xlo, Wkh, Wkl, bk, kh,  kl,  0);
  qkv_proj_kernel<<<pgrid, 128, 0, stream>>>(Xhi, Xlo, Wvh, Wvl, bv, vth, vtl, 1);

  // 3) attention: grid = (S/64 query groups, B*H)
  dim3 agrid(S / 64, Bsz * H);
  attn_kernel<<<agrid, 128, 0, stream>>>(qh, ql, kh, kl, vth, vtl, out);
}
